// MultiHeadAttention_71614284693616
// MI455X (gfx1250) — matrix operand order repaired and checked by reading, not compiled, not run on hardware
//
#include <hip/hip_runtime.h>
#include <hip/hip_bf16.h>

typedef __attribute__((ext_vector_type(16))) _Float16 v16h;
typedef __attribute__((ext_vector_type(8)))  _Float16 v8h;
typedef __attribute__((ext_vector_type(4)))  _Float16 v4h;
typedef __attribute__((ext_vector_type(8)))  float    v8f;

#define D_MODEL 1024
#define N_HEAD  16
#define D_HEAD  64
#define SEQ     2048
#define BATCH   4
#define M_TOTAL (BATCH * SEQ)   // 8192

// ---------------------------------------------------------------------------
// WMMA fragment helpers (wave32, v_wmma_f32_16x16x32_f16)
//
// A (16x32 f16, MxK)  lane m (<16): K 0..7 in v[0..3], K 16..23 in v[4..7]
//                     lane m+16   : K 8..15 / K 24..31        (ISA 7.12.2)
// B (32x16 f16, KxN)  lane n (<16): K 0..7 / K 16..23 of column n
//                     lane n+16   : K 8..15 / K 24..31 of column n
// C/D (16x16 f32)     vgpr r, lanes 0..15 -> M=r, lanes 16..31 -> M=r+8; N=lane&15
// ---------------------------------------------------------------------------

static __device__ __forceinline__ v16h load_frag_a(const _Float16* p, int ld, int lane) {
  const int r  = lane & 15;
  const int kk = (lane >> 4) * 8;
  const _Float16* q = p + (size_t)r * ld + kk;
  v8h lo = *(const v8h*)(q);        // K = kk .. kk+7
  v8h hi = *(const v8h*)(q + 16);   // K = kk+16 .. kk+23
  v16h out;
#pragma unroll
  for (int i = 0; i < 8; ++i) { out[i] = lo[i]; out[8 + i] = hi[i]; }
  return out;
}

// source rows indexed by N, K contiguous within a row (i.e. B stored N-major)
static __device__ __forceinline__ v16h load_frag_b(const _Float16* p, int ld, int lane) {
  const int n  = lane & 15;
  const int kk = (lane >> 4) * 8;
  const _Float16* q = p + (size_t)n * ld + kk;
  v8h lo = *(const v8h*)(q);        // K = kk .. kk+7
  v8h hi = *(const v8h*)(q + 16);   // K = kk+16 .. kk+23
  v16h out;
#pragma unroll
  for (int i = 0; i < 8; ++i) { out[i] = lo[i]; out[8 + i] = hi[i]; }
  return out;
}

static __device__ __forceinline__ v8f wmma_f16(v16h a, v16h b, v8f c) {
  return __builtin_amdgcn_wmma_f32_16x16x32_f16(false, a, false, b, (short)0, c,
                                                false, false);
}

// One k-step worth of operands for a 32x64 register tile
struct FragSet {
  v16h a0, a1;            // two 16-row A fragments
  v16h b0, b1, b2, b3;    // four 16-col B fragments
};

static __device__ __forceinline__ void load_set(FragSet& f,
    const _Float16* Xm0, const _Float16* Xm1,
    const _Float16* Wn0, const _Float16* Wn1,
    const _Float16* Wn2, const _Float16* Wn3,
    int k0, int lane) {
  f.a0 = load_frag_a(Xm0 + k0, D_MODEL, lane);
  f.a1 = load_frag_a(Xm1 + k0, D_MODEL, lane);
  f.b0 = load_frag_b(Wn0 + k0, D_MODEL, lane);
  f.b1 = load_frag_b(Wn1 + k0, D_MODEL, lane);
  f.b2 = load_frag_b(Wn2 + k0, D_MODEL, lane);
  f.b3 = load_frag_b(Wn3 + k0, D_MODEL, lane);
}

// 8 WMMAs: 2 A-fragments x 4 B-fragments (accumulate)
static __device__ __forceinline__ void mm8(v8f acc[2][4], const FragSet& f) {
  acc[0][0] = wmma_f16(f.a0, f.b0, acc[0][0]);
  acc[0][1] = wmma_f16(f.a0, f.b1, acc[0][1]);
  acc[0][2] = wmma_f16(f.a0, f.b2, acc[0][2]);
  acc[0][3] = wmma_f16(f.a0, f.b3, acc[0][3]);
  acc[1][0] = wmma_f16(f.a1, f.b0, acc[1][0]);
  acc[1][1] = wmma_f16(f.a1, f.b1, acc[1][1]);
  acc[1][2] = wmma_f16(f.a1, f.b2, acc[1][2]);
  acc[1][3] = wmma_f16(f.a1, f.b3, acc[1][3]);
}

// First k-step: C operand is a zeroinitializer -> lowers to inline 0 SRC2,
// avoiding 64 v_mov's of accumulator zero-init.
static __device__ __forceinline__ void mm8_first(v8f acc[2][4], const FragSet& f) {
  const v8f z = {};
  acc[0][0] = wmma_f16(f.a0, f.b0, z);
  acc[0][1] = wmma_f16(f.a0, f.b1, z);
  acc[0][2] = wmma_f16(f.a0, f.b2, z);
  acc[0][3] = wmma_f16(f.a0, f.b3, z);
  acc[1][0] = wmma_f16(f.a1, f.b0, z);
  acc[1][1] = wmma_f16(f.a1, f.b1, z);
  acc[1][2] = wmma_f16(f.a1, f.b2, z);
  acc[1][3] = wmma_f16(f.a1, f.b3, z);
}

// Ping-pong k-loop, no buffer-rotation copies, inline-zero first step.
// 32 k-steps total: prologue loads steps 0,1; loop covers 2..31; tail mm8.
static __device__ __forceinline__ void gemm_core(v8f acc[2][4],
    const _Float16* Xm0, const _Float16* Xm1,
    const _Float16* Wn0, const _Float16* Wn1,
    const _Float16* Wn2, const _Float16* Wn3, int lane) {
  FragSet f0, f1;
  load_set(f0, Xm0, Xm1, Wn0, Wn1, Wn2, Wn3, 0, lane);
  load_set(f1, Xm0, Xm1, Wn0, Wn1, Wn2, Wn3, 32, lane);
  mm8_first(acc, f0);
#pragma unroll 1
  for (int k0 = 64; k0 + 64 <= D_MODEL; k0 += 64) {
    __builtin_prefetch(Xm0 + k0 + 96, 0, 1);
    load_set(f0, Xm0, Xm1, Wn0, Wn1, Wn2, Wn3, k0, lane);
    mm8(acc, f1);
    load_set(f1, Xm0, Xm1, Wn0, Wn1, Wn2, Wn3, k0 + 32, lane);
    mm8(acc, f0);
  }
  mm8(acc, f1);   // last step (k = D_MODEL-32), loaded in final iteration
}

// ---------------------------------------------------------------------------
// fp32 -> f16 conversion
// ---------------------------------------------------------------------------
__global__ void __launch_bounds__(256) cvt_f32_to_f16(const float* __restrict__ src,
                                                      _Float16* __restrict__ dst, int n) {
  const int i = (blockIdx.x * blockDim.x + threadIdx.x) * 4;
  if (i >= n) return;
  const float4 v = *(const float4*)(src + i);
  v4h h;
  h[0] = (_Float16)v.x; h[1] = (_Float16)v.y;
  h[2] = (_Float16)v.z; h[3] = (_Float16)v.w;
  *(v4h*)(dst + i) = h;
}

// ---------------------------------------------------------------------------
// GEMM: C[m,n] = (sum_k X[m,k] * W[n,k] + bias[n]) * outScale   (W row-major [N,K])
// one wave per block; wave computes a 32x64 tile, ping-pong pipelined k-loop.
// vmode 0: write f16 [B,H,S,64]; vmode 1: write f16 [B,H,64,S] (transposed V)
// ---------------------------------------------------------------------------
__global__ void __launch_bounds__(32) gemm_qkv_kernel(
    const _Float16* __restrict__ X, const _Float16* __restrict__ W,
    const float* __restrict__ bias, _Float16* __restrict__ out,
    int vmode, float outScale) {
  const int lane = threadIdx.x;
  const int m0 = blockIdx.x * 32;
  const int n0 = blockIdx.y * 64;

  v8f acc[2][4];
  gemm_core(acc,
            X + (size_t)(m0 +  0) * D_MODEL, X + (size_t)(m0 + 16) * D_MODEL,
            W + (size_t)(n0 +  0) * D_MODEL, W + (size_t)(n0 + 16) * D_MODEL,
            W + (size_t)(n0 + 32) * D_MODEL, W + (size_t)(n0 + 48) * D_MODEL, lane);

  const int col = lane & 15;
  const int rb  = (lane >> 4) * 8;
#pragma unroll
  for (int t = 0; t < 2; ++t) {
#pragma unroll
    for (int j = 0; j < 4; ++j) {
      const int n = n0 + j * 16 + col;
      const float bv = bias[n];
      const int h = n >> 6;            // head
      const int d = n & (D_HEAD - 1);
#pragma unroll
      for (int r = 0; r < 8; ++r) {
        const int m  = m0 + 16 * t + rb + r;
        const int bb = m >> 11;        // / SEQ
        const int ss = m & (SEQ - 1);
        const _Float16 hv = (_Float16)((acc[t][j][r] + bv) * outScale);
        size_t idx;
        if (vmode == 0)
          idx = (((size_t)bb * N_HEAD + h) * SEQ + ss) * D_HEAD + d;
        else
          idx = (((size_t)bb * N_HEAD + h) * D_HEAD + d) * SEQ + ss;
        out[idx] = hv;
      }
    }
  }
}

// ---------------------------------------------------------------------------
// Final projection: out[m,n] = sum_k A[m,k] * Wo[n,k] + bo[n], fp32 output.
// ---------------------------------------------------------------------------
__global__ void __launch_bounds__(32) gemm_out_kernel(
    const _Float16* __restrict__ A, const _Float16* __restrict__ W,
    const float* __restrict__ bias, float* __restrict__ out) {
  const int lane = threadIdx.x;
  const int m0 = blockIdx.x * 32;
  const int n0 = blockIdx.y * 64;

  v8f acc[2][4];
  gemm_core(acc,
            A + (size_t)(m0 +  0) * D_MODEL, A + (size_t)(m0 + 16) * D_MODEL,
            W + (size_t)(n0 +  0) * D_MODEL, W + (size_t)(n0 + 16) * D_MODEL,
            W + (size_t)(n0 + 32) * D_MODEL, W + (size_t)(n0 + 48) * D_MODEL, lane);

  const int col = lane & 15;
  const int rb  = (lane >> 4) * 8;
#pragma unroll
  for (int t = 0; t < 2; ++t)
#pragma unroll
    for (int j = 0; j < 4; ++j) {
      const int n = n0 + j * 16 + col;
      const float bv = bias[n];
#pragma unroll
      for (int r = 0; r < 8; ++r) {
        const int m = m0 + 16 * t + rb + r;
        out[(size_t)m * D_MODEL + n] = acc[t][j][r] + bv;
      }
    }
}

// ---------------------------------------------------------------------------
// Flash-style attention: one wave handles 16 query rows of one (b,h).
// Q is pre-scaled by 1/sqrt(D_HEAD) in the projection epilogue, so scores
// come out of WMMA already scaled. Online softmax; O += P*V; normalize at end.
// K stored [B,H,S,64]; V stored transposed [B,H,64,S].
// Output written f16 into [B*S, D_MODEL] for the final projection GEMM.
// ---------------------------------------------------------------------------
__global__ void __launch_bounds__(32) attn_kernel(
    const _Float16* __restrict__ Q, const _Float16* __restrict__ K,
    const _Float16* __restrict__ Vt, _Float16* __restrict__ O) {
  __shared__ __align__(16) _Float16 Plds[16 * 32];

  const int lane = threadIdx.x;
  const int q0 = blockIdx.x * 16;
  const int h  = blockIdx.y;
  const int b  = blockIdx.z;

  const _Float16* Qb = Q  + ((size_t)b * N_HEAD + h) * SEQ * D_HEAD;
  const _Float16* Kb = K  + ((size_t)b * N_HEAD + h) * SEQ * D_HEAD;
  const _Float16* Vb = Vt + ((size_t)b * N_HEAD + h) * D_HEAD * SEQ;

  // Q fragments for d = 0..31 and 32..63 (constant over the kv loop)
  const v16h qa0 = load_frag_a(Qb + (size_t)q0 * D_HEAD + 0,  D_HEAD, lane);
  const v16h qa1 = load_frag_a(Qb + (size_t)q0 * D_HEAD + 32, D_HEAD, lane);

  float mrow[8], lrow[8];
  v8f o[4];
#pragma unroll
  for (int r = 0; r < 8; ++r) { mrow[r] = -3.0e38f; lrow[r] = 0.0f; }
#pragma unroll
  for (int j = 0; j < 4; ++j)
#pragma unroll
    for (int r = 0; r < 8; ++r) o[j][r] = 0.0f;

  const int col = lane & 15;
  const int rb  = (lane >> 4) * 8;

  // K fragments, double-buffered one kv-chunk ahead (loaded in place)
  v16h kb[4];
  kb[0] = load_frag_b(Kb + (size_t)0  * D_HEAD + 0,  D_HEAD, lane);
  kb[1] = load_frag_b(Kb + (size_t)0  * D_HEAD + 32, D_HEAD, lane);
  kb[2] = load_frag_b(Kb + (size_t)16 * D_HEAD + 0,  D_HEAD, lane);
  kb[3] = load_frag_b(Kb + (size_t)16 * D_HEAD + 32, D_HEAD, lane);

#pragma unroll 1
  for (int kv0 = 0; kv0 < SEQ; kv0 += 32) {
    // ---- scores: two 16x16 tiles over kv chunk of 32; first WMMA of each
    //      chain uses inline-zero C (no per-chunk accumulator re-init) ----
    const v8f z = {};
    v8f s0 = wmma_f16(qa0, kb[0], z);
    s0 = wmma_f16(qa1, kb[1], s0);
    v8f s1 = wmma_f16(qa0, kb[2], z);
    s1 = wmma_f16(qa1, kb[3], s1);

    // ---- issue V loads + next-chunk K loads before the softmax math ----
    v16h vb[4];
#pragma unroll
    for (int j = 0; j < 4; ++j)
      vb[j] = load_frag_b(Vb + (size_t)(j * 16) * SEQ + kv0, SEQ, lane);
    if (kv0 + 32 < SEQ) {
      const int kn = kv0 + 32;
      kb[0] = load_frag_b(Kb + (size_t)(kn +  0) * D_HEAD + 0,  D_HEAD, lane);
      kb[1] = load_frag_b(Kb + (size_t)(kn +  0) * D_HEAD + 32, D_HEAD, lane);
      kb[2] = load_frag_b(Kb + (size_t)(kn + 16) * D_HEAD + 0,  D_HEAD, lane);
      kb[3] = load_frag_b(Kb + (size_t)(kn + 16) * D_HEAD + 32, D_HEAD, lane);
    }

    // ---- online softmax (row stats replicated across the 16-lane half) -
#pragma unroll
    for (int r = 0; r < 8; ++r) {
      const float a = s0[r];
      const float c = s1[r];
      float v = fmaxf(a, c);
      v = fmaxf(v, __shfl_xor(v, 1, 32));
      v = fmaxf(v, __shfl_xor(v, 2, 32));
      v = fmaxf(v, __shfl_xor(v, 4, 32));
      v = fmaxf(v, __shfl_xor(v, 8, 32));
      const float mnew = fmaxf(mrow[r], v);
      const float f = __expf(mrow[r] - mnew);
      mrow[r] = mnew;
      const float p0 = __expf(a - mnew);
      const float p1 = __expf(c - mnew);
      s0[r] = p0; s1[r] = p1;
      float rs = p0 + p1;
      rs += __shfl_xor(rs, 1, 32);
      rs += __shfl_xor(rs, 2, 32);
      rs += __shfl_xor(rs, 4, 32);
      rs += __shfl_xor(rs, 8, 32);
      lrow[r] = lrow[r] * f + rs;
#pragma unroll
      for (int j = 0; j < 4; ++j) o[j][r] *= f;
    }

    // ---- re-layout P (C-layout -> A-layout) through LDS ----------------
    __syncthreads();
#pragma unroll
    for (int r = 0; r < 8; ++r) {
      Plds[(rb + r) * 32 + col]      = (_Float16)s0[r];
      Plds[(rb + r) * 32 + 16 + col] = (_Float16)s1[r];
    }
    __syncthreads();
    const v16h pa = load_frag_a(Plds, 32, lane);

    // ---- O += P * V  (V transposed: rows are d, kv contiguous) ---------
#pragma unroll
    for (int j = 0; j < 4; ++j)
      o[j] = wmma_f16(pa, vb[j], o[j]);
  }

  // ---- normalize and emit into [B*S, D_MODEL] f16 ----------------------
#pragma unroll
  for (int r = 0; r < 8; ++r) {
    const float inv = 1.0f / lrow[r];
    const size_t row = (size_t)b * SEQ + q0 + rb + r;
#pragma unroll
    for (int j = 0; j < 4; ++j) {
      O[row * D_MODEL + h * D_HEAD + j * 16 + col] = (_Float16)(o[j][r] * inv);
    }
  }
}

// ---------------------------------------------------------------------------
extern "C" void kernel_launch(void* const* d_in, const int* in_sizes, int n_in,
                              void* d_out, int out_size, void* d_ws, size_t ws_size,
                              hipStream_t stream) {
  const float* X  = (const float*)d_in[0];
  const float* Wq = (const float*)d_in[1];
  const float* bq = (const float*)d_in[2];
  const float* Wk = (const float*)d_in[3];
  const float* bk = (const float*)d_in[4];
  const float* Wv = (const float*)d_in[5];
  const float* bv = (const float*)d_in[6];
  const float* Wo = (const float*)d_in[7];
  const float* bo = (const float*)d_in[8];
  float* out = (float*)d_out;

  char* ws = (char*)d_ws;
  size_t off = 0;
  const size_t xBytes = (size_t)M_TOTAL * D_MODEL * sizeof(_Float16);   // 16 MB
  const size_t wBytes = (size_t)D_MODEL * D_MODEL * sizeof(_Float16);   //  2 MB
  _Float16* Xh  = (_Float16*)(ws + off); off += xBytes;
  _Float16* Wqh = (_Float16*)(ws + off); off += wBytes;
  _Float16* Wkh = (_Float16*)(ws + off); off += wBytes;
  _Float16* Wvh = (_Float16*)(ws + off); off += wBytes;
  _Float16* Woh = (_Float16*)(ws + off); off += wBytes;
  _Float16* Qh  = (_Float16*)(ws + off); off += xBytes;
  _Float16* Kh  = (_Float16*)(ws + off); off += xBytes;
  _Float16* Vth = (_Float16*)(ws + off); off += xBytes;
  _Float16* Oh  = (_Float16*)(ws + off); off += xBytes;   // total ~88 MB

  // ---- fp32 -> f16 conversions ----------------------------------------
  const int nX = M_TOTAL * D_MODEL;
  const int nW = D_MODEL * D_MODEL;
  cvt_f32_to_f16<<<nX / 4 / 256, 256, 0, stream>>>(X,  Xh,  nX);
  cvt_f32_to_f16<<<nW / 4 / 256, 256, 0, stream>>>(Wq, Wqh, nW);
  cvt_f32_to_f16<<<nW / 4 / 256, 256, 0, stream>>>(Wk, Wkh, nW);
  cvt_f32_to_f16<<<nW / 4 / 256, 256, 0, stream>>>(Wv, Wvh, nW);
  cvt_f32_to_f16<<<nW / 4 / 256, 256, 0, stream>>>(Wo, Woh, nW);

  // ---- QKV projections (Q carries the 1/sqrt(d_head) scale) ------------
  dim3 ggemm(M_TOTAL / 32, D_MODEL / 64);
  gemm_qkv_kernel<<<ggemm, 32, 0, stream>>>(Xh, Wqh, bq, Qh,  0, 0.125f);
  gemm_qkv_kernel<<<ggemm, 32, 0, stream>>>(Xh, Wkh, bk, Kh,  0, 1.0f);
  gemm_qkv_kernel<<<ggemm, 32, 0, stream>>>(Xh, Wvh, bv, Vth, 1, 1.0f);

  // ---- attention -------------------------------------------------------
  attn_kernel<<<dim3(SEQ / 16, N_HEAD, BATCH), 32, 0, stream>>>(Qh, Kh, Vth, Oh);

  // ---- output projection ----------------------------------------------
  gemm_out_kernel<<<ggemm, 32, 0, stream>>>(Oh, Woh, bo, out);
}
